// SAMAttention_40973988004726
// MI455X (gfx1250) — matrix operand order repaired and checked by reading, not compiled, not run on hardware
//
#include <hip/hip_runtime.h>

// Problem constants (from reference)
constexpr int kB = 2, kH = 16, kS = 2048, kD = 64;
constexpr float kScale = 0.125f;   // 1/sqrt(64)
constexpr float kMaskNeg = -1.0e9f;

typedef __attribute__((ext_vector_type(16))) _Float16 v16h;
typedef __attribute__((ext_vector_type(8)))  _Float16 v8h;
typedef __attribute__((ext_vector_type(4)))  _Float16 v4h;
typedef __attribute__((ext_vector_type(8)))  float    v8f;

static __device__ __forceinline__ v8f wmma_f16(v16h a, v16h b, v8f c) {
  // D = A(16x32 f16) * B(32x16 f16) + C(16x16 f32)
  return __builtin_amdgcn_wmma_f32_16x16x32_f16(false, a, false, b, (short)0, c,
                                                false, false);
}

// ---------------------------------------------------------------------------
// Pre-pass 1: elementwise f32 -> f16 convert (for K)
// ---------------------------------------------------------------------------
__global__ __launch_bounds__(256) void cvt_f32_f16(const float* __restrict__ src,
                                                   _Float16* __restrict__ dst) {
  size_t i = ((size_t)blockIdx.x * blockDim.x + threadIdx.x) * 4;
  float4 f = *(const float4*)(src + i);
  v4h h = { (_Float16)f.x, (_Float16)f.y, (_Float16)f.z, (_Float16)f.w };
  *(v4h*)(dst + i) = h;
}

// ---------------------------------------------------------------------------
// Pre-pass 2: V [b,h,s,d] f32 -> Vt [b,h,d,s] f16  (LDS-tiled transpose)
// ---------------------------------------------------------------------------
__global__ __launch_bounds__(256) void transpose_cvt_v(const float* __restrict__ v,
                                                       _Float16* __restrict__ vt) {
  __shared__ _Float16 t[64][66];           // pad to dodge bank conflicts
  const int bh = blockIdx.y;
  const int s0 = blockIdx.x * 64;
  const int tid = threadIdx.x;
  const float* vp = v + ((size_t)bh * kS + s0) * kD;
#pragma unroll
  for (int i = 0; i < 16; ++i) {
    int idx = i * 256 + tid;               // idx = sl*64 + d  (coalesced read)
    int sl = idx >> 6, d = idx & 63;
    t[d][sl] = (_Float16)vp[idx];
  }
  __syncthreads();
  _Float16* op = vt + (size_t)bh * kD * kS + s0;
#pragma unroll
  for (int i = 0; i < 16; ++i) {
    int idx = i * 256 + tid;               // idx = d*64 + sl  (coalesced write)
    int d = idx >> 6, sl = idx & 63;
    op[(size_t)d * kS + sl] = t[d][sl];
  }
}

// ---------------------------------------------------------------------------
// Main kernel: one wave32 per 16-query strip. 8 waves / block.
// Sweep A: row max.  Sweep B: sum of exp.  Sweep C: emit probs + PV.
// ---------------------------------------------------------------------------
__global__ __launch_bounds__(256) void sam_attn_kernel(
    const float* __restrict__ q, const _Float16* __restrict__ kf,
    const _Float16* __restrict__ vt, const float* __restrict__ cst,
    const int* __restrict__ mask, float* __restrict__ out,
    float* __restrict__ probs) {
  const int lane = threadIdx.x & 31;
  const int wave = threadIdx.x >> 5;
  const int nn = lane & 15;                // N / column index of fragments
  const int hh = lane >> 4;                // half-wave index

  const int strip = blockIdx.x * 8 + wave; // 16-query strip id
  const int bh = strip >> 7;               // strip / (S/16); 8 strips/block share bh
  const int q0 = (strip & 127) << 4;
  const int b = bh >> 4;
  const int h = bh & 15;

  const float ch = cst[h];
  const int* mrow = mask + b * kS;
  const _Float16* kp = kf + (size_t)bh * kS * kD;
  const _Float16* vtp = vt + (size_t)bh * kD * kS;

  // ---- Q A-fragments (16x32 f16 x2), f32->f16 in-register -----------------
  // A layout: lane row M = nn; elems 0..15 -> K = 16*hh+e
  const float* qrow = q + ((size_t)bh * kS + q0 + nn) * kD;
  v16h a0, a1;
#pragma unroll
  for (int p = 0; p < 8; ++p) {
    a0[p]     = (_Float16)qrow[16 * hh + p];
    a0[p + 8] = (_Float16)qrow[16 * hh + 8 + p];
    a1[p]     = (_Float16)qrow[32 + 16 * hh + p];
    a1[p + 8] = (_Float16)qrow[32 + 16 * hh + 8 + p];
  }

  // ---- Sweep A: row max (elementwise only; reduce once at the end) --------
  float rm[8];
#pragma unroll
  for (int r = 0; r < 8; ++r) rm[r] = -__builtin_inff();

  for (int k0 = 0; k0 < kS; k0 += 16) {
    const _Float16* krow = kp + (size_t)(k0 + nn) * kD;   // B: N=nn, K contiguous
    v16h b0 = *(const v16h*)(krow + 16 * hh);
    v16h b1 = *(const v16h*)(krow + 32 + 16 * hh);
    v8f c = {};
    c = wmma_f16(a0, b0, c);
    c = wmma_f16(a1, b1, c);
    const float mneg = mrow[k0 + nn] ? kMaskNeg : 0.0f;
#pragma unroll
    for (int r = 0; r < 8; ++r) rm[r] = fmaxf(rm[r], c[r] * kScale + mneg);
  }
#pragma unroll
  for (int r = 0; r < 8; ++r) {            // reduce across the 16-lane half
    float t = rm[r];
    t = fmaxf(t, __shfl_xor(t, 1));
    t = fmaxf(t, __shfl_xor(t, 2));
    t = fmaxf(t, __shfl_xor(t, 4));
    t = fmaxf(t, __shfl_xor(t, 8));
    rm[r] = t;
  }

  // ---- Sweep B: per-lane partial sum of exp(s - rm) -----------------------
  float ps[8];
#pragma unroll
  for (int r = 0; r < 8; ++r) ps[r] = 0.0f;

  for (int k0 = 0; k0 < kS; k0 += 16) {
    const _Float16* krow = kp + (size_t)(k0 + nn) * kD;
    v16h b0 = *(const v16h*)(krow + 16 * hh);
    v16h b1 = *(const v16h*)(krow + 32 + 16 * hh);
    v8f c = {};
    c = wmma_f16(a0, b0, c);
    c = wmma_f16(a1, b1, c);
    const float mneg = mrow[k0 + nn] ? kMaskNeg : 0.0f;
#pragma unroll
    for (int r = 0; r < 8; ++r)
      ps[r] += __expf(c[r] * kScale + mneg - rm[r]);
  }
  float invd[8];
#pragma unroll
  for (int r = 0; r < 8; ++r) {
    float t = ps[r];
    t += __shfl_xor(t, 1);
    t += __shfl_xor(t, 2);
    t += __shfl_xor(t, 4);
    t += __shfl_xor(t, 8);
    // leaky softmax denominator: sum + exp(c - max) * S
    invd[r] = 1.0f / (t + __expf(ch - rm[r]) * (float)kS);
  }

  // ---- Sweep C: emit probs (NT stream) + PV accumulation ------------------
  __shared__ __align__(64) _Float16 Pl[8][16][32];   // per-wave P bounce buffer
  _Float16(*P)[32] = Pl[wave];
  v8f o0 = {}, o1 = {}, o2 = {}, o3 = {};
  float* pr = probs + (size_t)bh * kS * kS + (size_t)(q0 + 8 * hh) * kS + nn;

  for (int k0 = 0; k0 < kS; k0 += 32) {
#pragma unroll
    for (int kt = 0; kt < 2; ++kt) {
      const int kk = k0 + kt * 16;
      const _Float16* krow = kp + (size_t)(kk + nn) * kD;
      v16h b0 = *(const v16h*)(krow + 16 * hh);
      v16h b1 = *(const v16h*)(krow + 32 + 16 * hh);
      v8f c = {};
      c = wmma_f16(a0, b0, c);
      c = wmma_f16(a1, b1, c);
      const float mneg = mrow[kk + nn] ? kMaskNeg : 0.0f;
#pragma unroll
      for (int r = 0; r < 8; ++r) {
        float p = __expf(c[r] * kScale + mneg - rm[r]) * invd[r];
        __builtin_nontemporal_store(p, pr + (size_t)r * kS + kk);
        P[r + 8 * hh][kt * 16 + nn] = (_Float16)p;  // C-layout -> row-major LDS
      }
    }
    // Re-gather P as A fragment (16x32): two contiguous 8-half runs per lane.
    v8h plo = *(const v8h*)(&P[nn][16 * hh]);
    v8h phi = *(const v8h*)(&P[nn][16 * hh + 8]);
    v16h pa = __builtin_shufflevector(plo, phi, 0, 1, 2, 3, 4, 5, 6, 7, 8, 9,
                                      10, 11, 12, 13, 14, 15);
    // V B fragments straight from transposed-f16 V: contiguous along S.
    const _Float16* vcol = vtp + (size_t)nn * kS + k0 + 16 * hh;
    o0 = wmma_f16(pa, *(const v16h*)(vcol), o0);
    o1 = wmma_f16(pa, *(const v16h*)(vcol + 16 * (size_t)kS), o1);
    o2 = wmma_f16(pa, *(const v16h*)(vcol + 32 * (size_t)kS), o2);
    o3 = wmma_f16(pa, *(const v16h*)(vcol + 48 * (size_t)kS), o3);
  }

  // ---- Write out tile -----------------------------------------------------
  float* orow = out + ((size_t)bh * kS + q0 + 8 * hh) * kD + nn;
#pragma unroll
  for (int r = 0; r < 8; ++r) {
    __builtin_nontemporal_store(o0[r], orow + (size_t)r * kD);
    __builtin_nontemporal_store(o1[r], orow + (size_t)r * kD + 16);
    __builtin_nontemporal_store(o2[r], orow + (size_t)r * kD + 32);
    __builtin_nontemporal_store(o3[r], orow + (size_t)r * kD + 48);
  }
}

// ---------------------------------------------------------------------------
extern "C" void kernel_launch(void* const* d_in, const int* in_sizes, int n_in,
                              void* d_out, int out_size, void* d_ws,
                              size_t ws_size, hipStream_t stream) {
  const float* q = (const float*)d_in[0];
  const float* k = (const float*)d_in[1];
  const float* v = (const float*)d_in[2];
  const float* cst = (const float*)d_in[3];
  const int* mask = (const int*)d_in[4];

  float* out = (float*)d_out;
  float* probs = out + (size_t)kB * kH * kS * kD;   // outputs concatenated

  // Workspace: K as f16 [b,h,s,d], V as f16 transposed [b,h,d,s] (16.8 MB)
  const size_t nel = (size_t)kB * kH * kS * kD;     // 4,194,304
  _Float16* kf16 = (_Float16*)d_ws;
  _Float16* vt16 = kf16 + nel;

  cvt_f32_f16<<<(unsigned)(nel / 4 / 256), 256, 0, stream>>>(k, kf16);
  transpose_cvt_v<<<dim3(kS / 64, kB * kH), 256, 0, stream>>>(v, vt16);
  sam_attn_kernel<<<(kB * kH * kS / 16) / 8, 256, 0, stream>>>(
      q, kf16, vt16, cst, mask, out, probs);
}
